// MultiHeadAttention2_87213605913183
// MI455X (gfx1250) — matrix operand order repaired and checked by reading, not compiled, not run on hardware
//
#include <hip/hip_runtime.h>
#include <cstdint>

// ---------------------------------------------------------------------------
// MHA forward for MI455X (gfx1250): bf16 WMMA GEMMs + single-pass fp32 softmax.
// Returns (output [B,S,D], attn_weights [B,H,S,S]) concatenated in d_out.
// ---------------------------------------------------------------------------

typedef __bf16  v16bf __attribute__((ext_vector_type(16)));
typedef __bf16  v8bf  __attribute__((ext_vector_type(8)));
typedef float   v8f   __attribute__((ext_vector_type(8)));

#define BDIM    4
#define SDIM    2048
#define DDIM    1024
#define HDIM    16
#define DKDIM   64
#define MDIM    (BDIM * SDIM)      // 8192 rows of activations
#define INV_SCALE 0.125f           // 1/sqrt(64)

__device__ __forceinline__ v16bf combine8(v8bf lo, v8bf hi) {
    v16bf r;
#pragma unroll
    for (int i = 0; i < 8; ++i) { r[i] = lo[i]; r[8 + i] = hi[i]; }
    return r;
}

__device__ __forceinline__ v16bf cvt2bf16(v8f lo, v8f hi) {
    v16bf r;
#pragma unroll
    for (int i = 0; i < 8; ++i) { r[i] = (__bf16)lo[i]; r[8 + i] = (__bf16)hi[i]; }
    return r;
}

// ---------------------------------------------------------------------------
// fp32 -> bf16 elementwise convert
// ---------------------------------------------------------------------------
__global__ void cvt_f32_bf16(const float* __restrict__ src,
                             __bf16* __restrict__ dst, int n) {
    int i = blockIdx.x * blockDim.x + threadIdx.x;
    if (i < n) dst[i] = (__bf16)src[i];
}

// ---------------------------------------------------------------------------
// Y[M,N] = A[M,K] * Bw[N,K]^T + bias[N]   (A,Bw bf16 row-major, K contiguous)
// Block: 256 thr = 8 waves. Each wave: 16 rows x 128 cols (8 WMMA tiles).
// Grid: (M/128, N/128).
// OUT_MODE: 0 = fp32 row-major [M,N], 1 = bf16 row-major [M,N],
//           2 = bf16 transposed-per-head Vt[b,h,d,s]  (N==DDIM, M==B*S)
// ---------------------------------------------------------------------------
template <int OUT_MODE>
__global__ __launch_bounds__(256) void gemm_bias_bf16(
    const __bf16* __restrict__ A, const __bf16* __restrict__ Bw,
    const float* __restrict__ bias,
    float* __restrict__ Cf, __bf16* __restrict__ Cb,
    int M, int N, int K)
{
    const int lane = threadIdx.x & 31;
    const int wave = threadIdx.x >> 5;
    const int rowBase = blockIdx.x * 128 + wave * 16;
    const int colBase = blockIdx.y * 128;
    const int m    = lane & 15;
    const int half = lane >> 4;

    v8f acc[8] = {};

    const __bf16* arow  = A  + (size_t)(rowBase + m) * K + half * 16;
    const __bf16* bbase = Bw + (size_t)(colBase + m) * K + half * 16;

    for (int kk = 0; kk < K; kk += 32) {
        // CDNA5 prefetch of the K-tile two steps ahead (speculative; dropped if OOB)
        __builtin_prefetch(arow + kk + 64, 0, 0);
        // Batch ALL fragment loads first so the compiler can clause them and
        // retire WMMAs against graduated s_wait_loadcnt values.
        v8bf alo = *(const v8bf*)(arow + kk);        // K = k0 .. k0+7
        v8bf ahi = *(const v8bf*)(arow + kk + 8);    // K = k0+8 .. k0+15
        v16bf bfrag[8];
#pragma unroll
        for (int t = 0; t < 8; ++t)
            bfrag[t] = *(const v16bf*)(bbase + (size_t)t * 16 * K + kk);
        v16bf afrag = combine8(alo, ahi);
#pragma unroll
        for (int t = 0; t < 8; ++t)
            acc[t] = __builtin_amdgcn_wmma_f32_16x16x32_bf16(
                false, afrag, false, bfrag[t], (short)0, acc[t], false, false);
    }

    const int mo = half * 8;
#pragma unroll
    for (int t = 0; t < 8; ++t) {
        const int cn = colBase + t * 16 + m;
        const float bv = bias[cn];
#pragma unroll
        for (int r = 0; r < 8; ++r) {
            const int   cm = rowBase + mo + r;            // global row (b*S + s)
            const float v  = acc[t][r] + bv;
            if (OUT_MODE == 0) {
                Cf[(size_t)cm * N + cn] = v;
            } else if (OUT_MODE == 1) {
                Cb[(size_t)cm * N + cn] = (__bf16)v;
            } else {
                // Vt[b,h,d,s] = val at row cm=(b*S+s), col cn=(h*64+d)
                const int b = cm >> 11;                   // / SDIM
                const int s = cm & (SDIM - 1);
                const int h = cn >> 6;                    // / DKDIM
                const int d = cn & (DKDIM - 1);
                Cb[(((size_t)b * HDIM + h) * DKDIM + d) * SDIM + s] = (__bf16)v;
            }
        }
    }
}

// ---------------------------------------------------------------------------
// scores[b,h,q,k] = sum_d Q[b,q,h*64+d] * K[b,k,h*64+d]   (raw, unscaled)
// Grid: (S/128, S/64, B*H).  Block: 256 thr = 8 waves (16 q-rows each).
// K-dim = 64: fully unrolled, all fragments loaded before the WMMA chain.
// ---------------------------------------------------------------------------
__global__ __launch_bounds__(256) void attn_scores(
    const __bf16* __restrict__ Q, const __bf16* __restrict__ Kb,
    float* __restrict__ Sc)
{
    const int bh = blockIdx.z;
    const int b  = bh / HDIM;
    const int h  = bh % HDIM;
    const int lane = threadIdx.x & 31;
    const int wave = threadIdx.x >> 5;
    const int qBase = blockIdx.x * 128 + wave * 16;
    const int kBase = blockIdx.y * 64;
    const int m    = lane & 15;
    const int half = lane >> 4;

    const __bf16* qrow  = Q  + ((size_t)b * SDIM + qBase + m) * DDIM + h * DKDIM + half * 16;
    const __bf16* kbase = Kb + ((size_t)b * SDIM + kBase + m) * DDIM + h * DKDIM + half * 16;

    // ---- load everything up front ----
    v16bf afr[2];
    v16bf bfr[2][4];
#pragma unroll
    for (int ks = 0; ks < 2; ++ks) {
        const int kk = ks * 32;
        v8bf alo = *(const v8bf*)(qrow + kk);
        v8bf ahi = *(const v8bf*)(qrow + kk + 8);
        afr[ks] = combine8(alo, ahi);
#pragma unroll
        for (int t = 0; t < 4; ++t)
            bfr[ks][t] = *(const v16bf*)(kbase + (size_t)t * 16 * DDIM + kk);
    }

    // ---- 8 WMMAs ----
    v8f acc[4] = {};
#pragma unroll
    for (int ks = 0; ks < 2; ++ks)
#pragma unroll
        for (int t = 0; t < 4; ++t)
            acc[t] = __builtin_amdgcn_wmma_f32_16x16x32_bf16(
                false, afr[ks], false, bfr[ks][t], (short)0, acc[t], false, false);

    const int mo = half * 8;
    float* out = Sc + (size_t)bh * SDIM * SDIM;
#pragma unroll
    for (int t = 0; t < 4; ++t) {
#pragma unroll
        for (int r = 0; r < 8; ++r) {
            out[(size_t)(qBase + mo + r) * SDIM + kBase + t * 16 + m] = acc[t][r];
        }
    }
}

// ---------------------------------------------------------------------------
// Single-pass in-place row softmax of exp((x - max)/SCALE) over rows of S.
// Row (2048) = 256 threads x 8 register-resident values: 1 read + 1 write.
// grid.x = B*H*S.
// ---------------------------------------------------------------------------
__global__ __launch_bounds__(256) void softmax_rows(float* __restrict__ P)
{
    float* p = P + (size_t)blockIdx.x * SDIM;
    __shared__ float red[8];
    const int tid  = threadIdx.x;
    const int lane = tid & 31;
    const int wave = tid >> 5;

    float v[8];
#pragma unroll
    for (int j = 0; j < 8; ++j) v[j] = p[tid + j * 256];

    // --- row max ---
    float mx = v[0];
#pragma unroll
    for (int j = 1; j < 8; ++j) mx = fmaxf(mx, v[j]);
#pragma unroll
    for (int o = 16; o > 0; o >>= 1) mx = fmaxf(mx, __shfl_xor(mx, o, 32));
    if (lane == 0) red[wave] = mx;
    __syncthreads();
    mx = red[0];
#pragma unroll
    for (int i = 1; i < 8; ++i) mx = fmaxf(mx, red[i]);
    __syncthreads();

    // --- exp in registers + row sum ---
    float sum = 0.0f;
#pragma unroll
    for (int j = 0; j < 8; ++j) {
        v[j] = __expf((v[j] - mx) * INV_SCALE);
        sum += v[j];
    }
#pragma unroll
    for (int o = 16; o > 0; o >>= 1) sum += __shfl_xor(sum, o, 32);
    if (lane == 0) red[wave] = sum;
    __syncthreads();
    sum = red[0];
#pragma unroll
    for (int i = 1; i < 8; ++i) sum += red[i];
    const float inv = 1.0f / sum;

#pragma unroll
    for (int j = 0; j < 8; ++j) p[tid + j * 256] = v[j] * inv;
}

// ---------------------------------------------------------------------------
// ctx[b,q,h*64+d] = sum_k P[b,h,q,k] * Vt[b,h,d,k]
// P is fp32 (read from d_out attn slice), converted to bf16 in-register.
// Grid: (S/128, 1, B*H).  Each wave: 16 q-rows x 64 d-cols, K-loop over S.
// ---------------------------------------------------------------------------
__global__ __launch_bounds__(256) void attn_pv(
    const float* __restrict__ P, const __bf16* __restrict__ Vt,
    __bf16* __restrict__ Ctx)
{
    const int bh = blockIdx.z;
    const int b  = bh / HDIM;
    const int h  = bh % HDIM;
    const int lane = threadIdx.x & 31;
    const int wave = threadIdx.x >> 5;
    const int qBase = blockIdx.x * 128 + wave * 16;
    const int m    = lane & 15;
    const int half = lane >> 4;

    v8f acc[4] = {};

    const float*  prow  = P  + ((size_t)bh * SDIM + qBase + m) * SDIM + half * 16;
    const __bf16* vbase = Vt + ((size_t)bh * DKDIM + m) * SDIM + half * 16;

    for (int kk = 0; kk < SDIM; kk += 32) {
        __builtin_prefetch(prow + kk + 64, 0, 0);   // P K-tile two steps ahead
        // batch all loads of this K-step, then convert + WMMA chain
        v8f plo = *(const v8f*)(prow + kk);
        v8f phi = *(const v8f*)(prow + kk + 8);
        v16bf bfrag[4];
#pragma unroll
        for (int t = 0; t < 4; ++t)
            bfrag[t] = *(const v16bf*)(vbase + (size_t)t * 16 * SDIM + kk);
        v16bf afrag = cvt2bf16(plo, phi);
#pragma unroll
        for (int t = 0; t < 4; ++t)
            acc[t] = __builtin_amdgcn_wmma_f32_16x16x32_bf16(
                false, afrag, false, bfrag[t], (short)0, acc[t], false, false);
    }

    const int mo = half * 8;
#pragma unroll
    for (int t = 0; t < 4; ++t) {
#pragma unroll
        for (int r = 0; r < 8; ++r) {
            Ctx[((size_t)b * SDIM + qBase + mo + r) * DDIM + h * DKDIM + t * 16 + m]
                = (__bf16)acc[t][r];
        }
    }
}

// ---------------------------------------------------------------------------
// Launcher
// ---------------------------------------------------------------------------
extern "C" void kernel_launch(void* const* d_in, const int* in_sizes, int n_in,
                              void* d_out, int out_size, void* d_ws, size_t ws_size,
                              hipStream_t stream) {
    const float* qin = (const float*)d_in[0];
    const float* kin = (const float*)d_in[1];
    const float* vin = (const float*)d_in[2];
    const float* Wq  = (const float*)d_in[3];
    const float* bq  = (const float*)d_in[4];
    const float* Wk  = (const float*)d_in[5];
    const float* bk  = (const float*)d_in[6];
    const float* Wv  = (const float*)d_in[7];
    const float* bv  = (const float*)d_in[8];
    const float* Wo  = (const float*)d_in[9];
    const float* bo  = (const float*)d_in[10];

    float* outP  = (float*)d_out;                              // [B,S,D]
    float* attnP = (float*)d_out + (size_t)MDIM * DDIM;        // [B,H,S,S]

    // ---- carve workspace (bf16 buffers) ----
    const size_t nAct = (size_t)MDIM * DDIM;   // 8192*1024
    const size_t nW   = (size_t)DDIM * DDIM;   // 1024*1024
    char* ws = (char*)d_ws;
    size_t off = 0;
    auto carve = [&](size_t elems) {
        void* p = ws + off;
        off += ((elems * sizeof(__bf16)) + 255) & ~(size_t)255;
        return (__bf16*)p;
    };
    __bf16* Xq  = carve(nAct);  __bf16* Xk  = carve(nAct);  __bf16* Xv = carve(nAct);
    __bf16* Wqb = carve(nW);    __bf16* Wkb = carve(nW);
    __bf16* Wvb = carve(nW);    __bf16* Wob = carve(nW);
    __bf16* Qb  = carve(nAct);  __bf16* Kb  = carve(nAct);
    __bf16* Vt  = carve(nAct);                 // V projection stored transposed
    __bf16* Ctx = carve(nAct);

    // ---- 1) fp32 -> bf16 converts ----
    {
        const int T = 256;
        cvt_f32_bf16<<<(int)((nAct + T - 1) / T), T, 0, stream>>>(qin, Xq, (int)nAct);
        cvt_f32_bf16<<<(int)((nAct + T - 1) / T), T, 0, stream>>>(kin, Xk, (int)nAct);
        cvt_f32_bf16<<<(int)((nAct + T - 1) / T), T, 0, stream>>>(vin, Xv, (int)nAct);
        cvt_f32_bf16<<<(int)((nW   + T - 1) / T), T, 0, stream>>>(Wq,  Wqb, (int)nW);
        cvt_f32_bf16<<<(int)((nW   + T - 1) / T), T, 0, stream>>>(Wk,  Wkb, (int)nW);
        cvt_f32_bf16<<<(int)((nW   + T - 1) / T), T, 0, stream>>>(Wv,  Wvb, (int)nW);
        cvt_f32_bf16<<<(int)((nW   + T - 1) / T), T, 0, stream>>>(Wo,  Wob, (int)nW);
    }

    // ---- 2) projections: Q/K = X @ W^T + b (bf16), V directly transposed ----
    {
        dim3 grid(MDIM / 128, DDIM / 128);
        gemm_bias_bf16<1><<<grid, 256, 0, stream>>>(Xq, Wqb, bq, nullptr, Qb,
                                                    MDIM, DDIM, DDIM);
        gemm_bias_bf16<1><<<grid, 256, 0, stream>>>(Xk, Wkb, bk, nullptr, Kb,
                                                    MDIM, DDIM, DDIM);
        gemm_bias_bf16<2><<<grid, 256, 0, stream>>>(Xv, Wvb, bv, nullptr, Vt,
                                                    MDIM, DDIM, DDIM);
    }

    // ---- 3) raw scores into attn slice of d_out ----
    {
        dim3 grid(SDIM / 128, SDIM / 64, BDIM * HDIM);
        attn_scores<<<grid, 256, 0, stream>>>(Qb, Kb, attnP);
    }

    // ---- 4) scaled softmax in place (this IS the attn_weights output) ----
    softmax_rows<<<BDIM * HDIM * SDIM, 256, 0, stream>>>(attnP);

    // ---- 5) PV ----
    {
        dim3 grid(SDIM / 128, 1, BDIM * HDIM);
        attn_pv<<<grid, 256, 0, stream>>>(attnP, Vt, Ctx);
    }

    // ---- 6) output projection: out = ctx @ Wo^T + bo  (fp32 out) ----
    {
        dim3 grid(MDIM / 128, DDIM / 128);
        gemm_bias_bf16<0><<<grid, 256, 0, stream>>>(Ctx, Wob, bo, outP, nullptr,
                                                    MDIM, DDIM, DDIM);
    }
}
